// Hybrid_IIN_18322330484876
// MI455X (gfx1250) — matrix operand order repaired and checked by reading, not compiled, not run on hardware
//
#include <hip/hip_runtime.h>
#include <hip/hip_bf16.h>

typedef __attribute__((ext_vector_type(16))) _Float16 v16h;
typedef __attribute__((ext_vector_type(8)))  _Float16 v8h;
typedef __attribute__((ext_vector_type(8)))  float    v8f;
typedef __attribute__((address_space(3)))    _Float16 lds_f16;

constexpr int Bn = 256, Tn = 512, FVn = 64, Hn = 64, Cn = 32, En = 64, NHn = 2;
constexpr long BTn = (long)Bn * Tn;        // 131072 rows
constexpr long Sn  = BTn * FVn;            // 8388608 elements per output tensor
constexpr int NBLK_LOSS = (int)(Sn / 256); // 32768

// ---------------- device helpers -------------------------------------------

__device__ __forceinline__ float fast_rcp(float x) { return __builtin_amdgcn_rcpf(x); }
// sigmoid: v_exp + v_rcp, no IEEE-division expansion, saturates correctly
__device__ __forceinline__ float sigm(float x) { return fast_rcp(1.f + __expf(-x)); }
// tanh(x) = 1 - 2/(1+exp(2x)): branch-free, exp saturation gives +-1 at the ends
__device__ __forceinline__ float tanh_fast(float x) {
  return 1.f - 2.f * fast_rcp(1.f + __expf(2.f * x));
}

// A fragment: 16x32 f16, rows = M (lane&15), K contiguous per lane-half:
// lane-half 0 -> K in {0..15}, half 1 -> K in {16..31}
__device__ __forceinline__ v16h ldfragA(const _Float16* base, int ld) {
  int lane = threadIdx.x & 31;
  int r  = lane & 15;
  int kb = (lane >> 4) << 4;              // 0 or 16
  const _Float16* p = base + (long)r * ld + kb;
  v8h lo = *(const v8h*)(p);
  v8h hi = *(const v8h*)(p + 8);
  v16h v;
#pragma unroll
  for (int j = 0; j < 8; ++j) { v[j] = lo[j]; v[8 + j] = hi[j]; }
  return v;
}

// B fragment from an N x K row-major matrix (i.e. W when computing X @ W^T).
// K order matches ldfragA: lane = N, lane-half g -> K = g*16 + j (contiguous).
__device__ __forceinline__ v16h ldfragB_nk(const _Float16* base, int ld) {
  int lane = threadIdx.x & 31;
  int r  = lane & 15;
  int kb = (lane >> 4) << 4;              // 0 or 16
  const _Float16* p = base + (long)r * ld + kb;
  v8h lo = *(const v8h*)(p);
  v8h hi = *(const v8h*)(p + 8);
  v16h v;
#pragma unroll
  for (int j = 0; j < 8; ++j) { v[j] = lo[j]; v[8 + j] = hi[j]; }
  return v;
}

// B fragment from a K x N row-major matrix (e.g. V in attention P@V).
__device__ __forceinline__ v16h ldfragB_kn(const _Float16* base, int ld) {
  int lane = threadIdx.x & 31;
  int n  = lane & 15;
  int kb = (lane >> 4) << 4;
  v16h v;
#pragma unroll
  for (int j = 0; j < 16; ++j) v[j] = base[(long)(kb + j) * ld + n];
  return v;
}

__device__ __forceinline__ v8f wmma16(v16h a, v16h b, v8f c) {
  return __builtin_amdgcn_wmma_f32_16x16x32_f16(false, a, false, b, (short)0, c,
                                                false, false);
}

// ---------------- small utility kernels ------------------------------------

__global__ void f32_to_f16_k(const float* __restrict__ s, _Float16* __restrict__ d, int n) {
  int i = blockIdx.x * 256 + threadIdx.x;
  if (i < n) d[i] = (_Float16)s[i];
}

__global__ void addvec_k(const float* __restrict__ a, const float* __restrict__ b,
                         float* __restrict__ d, int n) {
  int i = blockIdx.x * 256 + threadIdx.x;
  if (i < n) d[i] = a[i] + b[i];
}

__global__ void tanh_k(const float* __restrict__ s, float* __restrict__ d, int n) {
  int i = blockIdx.x * 256 + threadIdx.x;
  if (i < n) d[i] = tanhf(s[i]);
}

// per-(b,f): data means -> normalized means, (max-min), and 1/(max-min)
__global__ void stats_k(const float* __restrict__ values, const float* __restrict__ masks,
                        const float* __restrict__ minv, const float* __restrict__ maxv,
                        float* __restrict__ mmn, float* __restrict__ rmn,
                        float* __restrict__ nmeans) {
  int b = blockIdx.x, f = threadIdx.x;
  float sv = 0.f, sm = 0.f;
  const float* vp = values + ((long)b * Tn) * FVn + f;
  const float* mp = masks  + ((long)b * Tn) * FVn + f;
  for (int t = 0; t < Tn; ++t) { sv += vp[(long)t * FVn]; sm += mp[(long)t * FVn]; }
  float mn = minv[b * FVn + f], mx = maxv[b * FVn + f];
  float d = mx - mn;
  mmn[b * FVn + f]    = d;
  rmn[b * FVn + f]    = 1.f / d;          // precise divide, done once per (b,f)
  nmeans[b * FVn + f] = (sv / sm - mn) / d;
}

// decay + compositing: writes xs_h = [x_comp | deltas] (f16, BT x 128)
__global__ void preproc_k(const float* __restrict__ values, const float* __restrict__ masks,
                          const float* __restrict__ deltas, const float* __restrict__ minv,
                          const float* __restrict__ rmn, const float* __restrict__ nmeans,
                          const float* __restrict__ decay_W, const float* __restrict__ decay_b,
                          const int* __restrict__ lengths, _Float16* __restrict__ xs_h) {
  long idx = (long)blockIdx.x * 256 + threadIdx.x;
  if (idx >= Sn) return;
  int f = (int)(idx & 63);
  long row = idx >> 6;
  int t = (int)(row & (Tn - 1));
  int b = (int)(row >> 9);
  float mn = minv[b * FVn + f], rd = rmn[b * FVn + f];
  float nv = (values[idx] - mn) * rd;
  long pidx = (row - (t > 0 ? 1 : 0)) * FVn + f;
  float vprev = (values[pidx] - mn) * rd;
  float g  = __expf(-fmaxf(deltas[idx] * decay_W[f * FVn + f] + decay_b[f], 0.f));
  float xd = g * vprev + (1.f - g) * nmeans[b * FVn + f];
  float m  = masks[idx];
  float pm = ((t + 1) <= lengths[b]) ? 1.f : 0.f;
  float xc = (m * nv + (1.f - m) * xd) * pm;
  xs_h[row * 128 + f]       = (_Float16)xc;
  xs_h[row * 128 + 64 + f]  = (_Float16)deltas[idx];
}

// li = [x_comp | masks | ctx(b)]  (f16, BT x 160)
__global__ void build_li_k(const _Float16* __restrict__ xs_h, const float* __restrict__ masks,
                           const _Float16* __restrict__ ctxh, _Float16* __restrict__ li_h) {
  long idx = (long)blockIdx.x * 256 + threadIdx.x;
  if (idx >= BTn * 160) return;
  int c = (int)(idx % 160);
  long row = idx / 160;
  int b = (int)(row >> 9);
  _Float16 v;
  if (c < 64)       v = xs_h[row * 128 + c];
  else if (c < 128) v = (_Float16)masks[row * FVn + (c - 64)];
  else              v = ctxh[b * Cn + (c - 128)];
  li_h[row * 160 + c] = v;
}

// ---------------- generic WMMA GEMM: D = A(f16) @ W(f16)^T + bias ----------
// A: rows x K (row-major, f16), W: N x K (row-major, f16).
// Each wave computes a 16 x (16*NT) output panel, reusing one A fragment
// across NT column tiles (cuts A HBM/L2 traffic NT-fold).
template <int NT>
__global__ __launch_bounds__(32) void gemm16_k(const _Float16* __restrict__ A, int lda,
                                               const _Float16* __restrict__ W, int ldw,
                                               const float* __restrict__ bias,
                                               float* __restrict__ out32, int ldo32,
                                               _Float16* __restrict__ out16, int ldo16,
                                               int K, int act) {
  long r0 = (long)blockIdx.x * 16;
  int  n0 = blockIdx.y * (16 * NT);
  v8f acc[NT];
#pragma unroll
  for (int i = 0; i < NT; ++i) acc[i] = (v8f){};
  for (int k = 0; k < K; k += 32) {
    v16h a = ldfragA(A + r0 * lda + k, lda);
    __builtin_prefetch(A + r0 * lda + k + 32, 0, 1);  // next K chunk
#pragma unroll
    for (int i = 0; i < NT; ++i) {
      v16h b = ldfragB_nk(W + (long)(n0 + 16 * i) * ldw + k, ldw);
      acc[i] = wmma16(a, b, acc[i]);
    }
  }
  int lane = threadIdx.x;
  int n = lane & 15, hg = lane >> 4;
#pragma unroll
  for (int i = 0; i < NT; ++i) {
    int nc = n0 + 16 * i + n;
    float bb = bias ? bias[nc] : 0.f;
#pragma unroll
    for (int v = 0; v < 8; ++v) {
      float x = acc[i][v] + bb;
      if (act == 1) x = fmaxf(x, 0.f);
      long m = r0 + v + 8 * hg;
      if (out32) out32[m * ldo32 + nc] = x;
      if (out16) out16[m * ldo16 + nc] = (_Float16)x;
    }
  }
}

// ---------------- GRU scan (16 batch rows / block, WMMA recurrent) ---------
__global__ __launch_bounds__(32) void gru_scan_k(const float* __restrict__ gi,
                                                 const _Float16* __restrict__ whh,
                                                 const float* __restrict__ bh,
                                                 const int* __restrict__ lengths,
                                                 _Float16* __restrict__ ctxh) {
  __shared__ __align__(16) _Float16 hsh[16 * 32];
  __shared__ float hf[16 * 32];
  __shared__ float hn[16 * 32];
  __shared__ int ls[16];
  int lane = threadIdx.x;
  int b0 = blockIdx.x * 16;
  for (int i = lane; i < 16 * 32; i += 32) { hsh[i] = (_Float16)0.f; hf[i] = 0.f; hn[i] = 0.f; }
  if (lane < 16) ls[lane] = lengths[b0 + lane];
  v16h wb[6];
#pragma unroll
  for (int i = 0; i < 6; ++i) wb[i] = ldfragB_nk(whh + i * 16 * 32, 32);
  int n = lane & 15, hg = lane >> 4;
  __syncthreads();
  for (int t = 0; t < Tn; ++t) {
    v16h a = ldfragA(hsh, 32);
    v8f z = {};
    v8f d0 = wmma16(a, wb[0], z);
    v8f d1 = wmma16(a, wb[1], z);
    v8f d2 = wmma16(a, wb[2], z);
    v8f d3 = wmma16(a, wb[3], z);
    v8f d4 = wmma16(a, wb[4], z);
    v8f d5 = wmma16(a, wb[5], z);
    __syncthreads();
#pragma unroll
    for (int v = 0; v < 8; ++v) {
      int m = v + 8 * hg;
      const float* gr = gi + ((long)(b0 + m) * Tn + t) * 96;
      __builtin_prefetch(gr + 96, 0, 1);  // next timestep's gate row
      float mt = ((t + 1) <= ls[m]) ? 1.f : 0.f;
      {
        int c = n;
        float rg = sigm(gr[c] + d0[v] + bh[c]);
        float zg = sigm(gr[32 + c] + d2[v] + bh[32 + c]);
        float ng = tanh_fast(gr[64 + c] + rg * (d4[v] + bh[64 + c]));
        float hnew = (1.f - zg) * ng + zg * hf[m * 32 + c];
        hf[m * 32 + c] = hnew; hsh[m * 32 + c] = (_Float16)hnew;
        hn[m * 32 + c] = mt * hnew + (1.f - mt) * hn[m * 32 + c];
      }
      {
        int c = n + 16;
        float rg = sigm(gr[c] + d1[v] + bh[c]);
        float zg = sigm(gr[32 + c] + d3[v] + bh[32 + c]);
        float ng = tanh_fast(gr[64 + c] + rg * (d5[v] + bh[64 + c]));
        float hnew = (1.f - zg) * ng + zg * hf[m * 32 + c];
        hf[m * 32 + c] = hnew; hsh[m * 32 + c] = (_Float16)hnew;
        hn[m * 32 + c] = mt * hnew + (1.f - mt) * hn[m * 32 + c];
      }
    }
    __syncthreads();
  }
  for (int i = lane; i < 16 * 32; i += 32) {
    int m = i >> 5, c = i & 31;
    ctxh[(b0 + m) * 32 + c] = (_Float16)hn[i];
  }
}

// ---------------- LSTM scan (fwd/bwd), WMMA recurrent h @ wh^T -------------
__global__ __launch_bounds__(32) void lstm_scan_k(const _Float16* __restrict__ gih,
                                                  const _Float16* __restrict__ whh,
                                                  const float* __restrict__ h0,
                                                  const float* __restrict__ c0,
                                                  _Float16* __restrict__ comb,
                                                  int backward) {
  __shared__ __align__(16) _Float16 whl[256 * 64];
  __shared__ __align__(16) _Float16 hsh[16 * 64];
  __shared__ float hstate[16 * 64];
  __shared__ float cstate[16 * 64];
  __shared__ float gbuf[16 * 256];
  int lane = threadIdx.x;
  int b0 = blockIdx.x * 16;
  // Stage the 32KB recurrent weight panel via ASYNCcnt-tracked async copies
  // (global_load_async_to_lds_b128, tracked with s_wait_asynccnt).
  {
    unsigned lbase = (unsigned)(unsigned long long)(lds_f16*)whl;
    for (int chunk = lane; chunk < (256 * 64) / 8; chunk += 32) {  // 16B chunks
      unsigned loff = lbase + (unsigned)chunk * 16u;
      const _Float16* g = whh + (long)chunk * 8;
      asm volatile("global_load_async_to_lds_b128 %0, %1, off"
                   :: "v"(loff), "v"(g) : "memory");
    }
    asm volatile("s_wait_asynccnt 0x0" ::: "memory");
  }
  for (int i = lane; i < 16 * 64; i += 32) {
    int m = i >> 6, c = i & 63;
    float hv = h0[(b0 + m) * 64 + c];
    hstate[i] = hv; hsh[i] = (_Float16)hv;
    cstate[i] = c0[(b0 + m) * 64 + c];
  }
  int n = lane & 15, hg = lane >> 4;
  __syncthreads();
  for (int s = 0; s < Tn; ++s) {
    int t = backward ? (Tn - 1 - s) : s;
    v16h a0 = ldfragA(hsh, 64);
    v16h a1 = ldfragA(hsh + 32, 64);
#pragma unroll 1
    for (int tile = 0; tile < 16; ++tile) {
      v16h b0f = ldfragB_nk(whl + tile * 16 * 64, 64);
      v16h b1f = ldfragB_nk(whl + tile * 16 * 64 + 32, 64);
      v8f acc = {};
      acc = wmma16(a0, b0f, acc);
      acc = wmma16(a1, b1f, acc);
#pragma unroll
      for (int v = 0; v < 8; ++v)
        gbuf[(v + 8 * hg) * 256 + tile * 16 + n] = acc[v];
    }
    __syncthreads();
#pragma unroll
    for (int v = 0; v < 8; ++v) {
      int m = v + 8 * hg;
      long grow = ((long)(b0 + m) * Tn + t) * 256;
      __builtin_prefetch(gih + grow + (backward ? -256 : 256), 0, 1);
#pragma unroll
      for (int s2 = 0; s2 < 4; ++s2) {
        int c = n + 16 * s2;
        float ig = (float)gih[grow + c]        + gbuf[m * 256 + c];
        float fg = (float)gih[grow + 64 + c]   + gbuf[m * 256 + 64 + c];
        float gg = (float)gih[grow + 128 + c]  + gbuf[m * 256 + 128 + c];
        float og = (float)gih[grow + 192 + c]  + gbuf[m * 256 + 192 + c];
        float cn = sigm(fg) * cstate[m * 64 + c] + sigm(ig) * tanh_fast(gg);
        float hv = sigm(og) * tanh_fast(cn);
        cstate[m * 64 + c] = cn;
        hstate[m * 64 + c] = hv;
        hsh[m * 64 + c] = (_Float16)hv;
        comb[((long)(b0 + m) * Tn + t) * 128 + c] = (_Float16)hv;
      }
    }
    __syncthreads();
  }
}

// ---------------- fused attention per (b, head, 16-query tile) -------------
__global__ __launch_bounds__(32) void attn_k(const _Float16* __restrict__ qkv,
                                             _Float16* __restrict__ ao) {
  __shared__ float sc[16 * 512];
  __shared__ __align__(16) _Float16 pb[16 * 512];
  int qt = blockIdx.x, hh = blockIdx.y, b = blockIdx.z;
  int lane = threadIdx.x, n = lane & 15, hg = lane >> 4;
  const float scale = 0.17677669529663687f;  // 1/sqrt(32)
  v16h a = ldfragA(qkv + ((long)b * Tn + qt * 16) * 192 + hh * 32, 192);
  for (int kt = 0; kt < 32; ++kt) {
    v16h bf = ldfragB_nk(qkv + ((long)b * Tn + kt * 16) * 192 + 64 + hh * 32, 192);
    v8f z = {};
    v8f d = wmma16(a, bf, z);
#pragma unroll
    for (int v = 0; v < 8; ++v)
      sc[(v + 8 * hg) * 512 + kt * 16 + n] = d[v] * scale;
  }
  __syncthreads();
  {
    // full-wave softmax: each lane handles half of one row, combine via shfl
    float* row = sc + n * 512 + hg * 256;
    float mx = -1e30f;
    for (int j = 0; j < 256; ++j) mx = fmaxf(mx, row[j]);
    mx = fmaxf(mx, __shfl_xor(mx, 16, 32));
    float sum = 0.f;
    for (int j = 0; j < 256; ++j) { float e = __expf(row[j] - mx); row[j] = e; sum += e; }
    sum += __shfl_xor(sum, 16, 32);
    float inv = fast_rcp(sum);
    _Float16* prow = pb + n * 512 + hg * 256;
    for (int j = 0; j < 256; ++j) prow[j] = (_Float16)(row[j] * inv);
  }
  __syncthreads();
#pragma unroll 1
  for (int nt = 0; nt < 2; ++nt) {
    v8f acc = {};
    for (int kc = 0; kc < 16; ++kc) {
      v16h pa = ldfragA(pb + kc * 32, 512);
      v16h vb = ldfragB_kn(qkv + ((long)b * Tn + kc * 32) * 192 + 128 + hh * 32 + nt * 16, 192);
      acc = wmma16(pa, vb, acc);
    }
#pragma unroll
    for (int v = 0; v < 8; ++v)
      ao[((long)b * Tn + qt * 16 + v + 8 * hg) * 64 + hh * 32 + nt * 16 + n] = (_Float16)acc[v];
  }
}

// ---------------- gating head (N=1) ----------------------------------------
__global__ void beta_k(const _Float16* __restrict__ fx2, const float* __restrict__ gw,
                       const float* __restrict__ gb, float* __restrict__ beta) {
  long row = (long)blockIdx.x * 256 + threadIdx.x;
  if (row >= BTn) return;
  float s = gb[0];
  for (int j = 0; j < 64; ++j) s += (float)fx2[row * 64 + j] * gw[j];
  beta[row] = sigm(s);
}

// ---------------- fusion + rescale + loss partials -------------------------
__global__ void final_k(const float* __restrict__ values, const float* __restrict__ masks,
                        const float* __restrict__ minv, const float* __restrict__ mmn,
                        const float* __restrict__ rmn,
                        const float* __restrict__ feat, const float* __restrict__ rnn,
                        const float* __restrict__ beta, const int* __restrict__ lengths,
                        float* __restrict__ out, float* __restrict__ partial) {
  __shared__ float red[256];
  __shared__ float redm[256];
  long idx = (long)blockIdx.x * 256 + threadIdx.x;
  float l = 0.f, msum = 0.f;
  if (idx < Sn) {
    int f = (int)(idx & 63);
    long row = idx >> 6;
    int t = (int)(row & (Tn - 1));
    int b = (int)(row >> 9);
    float mn = minv[b * 64 + f], d = mmn[b * 64 + f];
    float nv = (values[idx] - mn) * rmn[b * 64 + f];
    float m = masks[idx];
    float ft = feat[row * 64 + f];
    float rn = rnn[row * 64 + f];
    float be = beta[row];
    float imp = be * ft + (1.f - be) * rn;
    float fin = m * nv + (1.f - m) * imp;
    float pm = ((t + 1) <= lengths[b]) ? 1.f : 0.f;
    out[idx]            = pm != 0.f ? fin * d + mn : 0.f;
    out[Sn + idx]       = pm != 0.f ? ft * d + mn : 0.f;
    out[2 * Sn + idx]   = pm != 0.f ? rn * d + mn : 0.f;
    l = fabsf(rn * m - nv * m) + fabsf(ft * m - nv * m) + fabsf(imp * m - nv * m);
    msum = m;
  }
  red[threadIdx.x] = l; redm[threadIdx.x] = msum;
  __syncthreads();
  for (int s = 128; s > 0; s >>= 1) {
    if (threadIdx.x < (unsigned)s) {
      red[threadIdx.x]  += red[threadIdx.x + s];
      redm[threadIdx.x] += redm[threadIdx.x + s];
    }
    __syncthreads();
  }
  if (threadIdx.x == 0) {
    partial[blockIdx.x] = red[0];
    partial[NBLK_LOSS + blockIdx.x] = redm[0];
  }
}

__global__ void loss_finalize_k(const float* __restrict__ partial, float* __restrict__ out) {
  __shared__ float r0[256];
  __shared__ float r1[256];
  float a = 0.f, b = 0.f;
  for (int i = threadIdx.x; i < NBLK_LOSS; i += 256) {
    a += partial[i];
    b += partial[NBLK_LOSS + i];
  }
  r0[threadIdx.x] = a; r1[threadIdx.x] = b;
  __syncthreads();
  for (int s = 128; s > 0; s >>= 1) {
    if (threadIdx.x < (unsigned)s) {
      r0[threadIdx.x] += r0[threadIdx.x + s];
      r1[threadIdx.x] += r1[threadIdx.x + s];
    }
    __syncthreads();
  }
  if (threadIdx.x == 0) out[3 * Sn] = r0[0] / r1[0];
}

// ---------------- host launcher --------------------------------------------

extern "C" void kernel_launch(void* const* d_in, const int* in_sizes, int n_in,
                              void* d_out, int out_size, void* d_ws, size_t ws_size,
                              hipStream_t stream) {
  (void)in_sizes; (void)n_in; (void)out_size; (void)ws_size;
  const float* values   = (const float*)d_in[0];
  const float* masks    = (const float*)d_in[1];
  const float* deltas   = (const float*)d_in[2];
  const float* min_vals = (const float*)d_in[3];
  const float* max_vals = (const float*)d_in[4];
  const float* decay_W  = (const float*)d_in[5];
  const float* decay_b  = (const float*)d_in[6];
  const float* gru_wi   = (const float*)d_in[7];
  const float* gru_wh   = (const float*)d_in[8];
  const float* gru_bi   = (const float*)d_in[9];
  const float* gru_bh   = (const float*)d_in[10];
  const float* ih_w     = (const float*)d_in[11];
  const float* ih_b     = (const float*)d_in[12];
  const float* lf_wi    = (const float*)d_in[13];
  const float* lf_wh    = (const float*)d_in[14];
  const float* lf_bi    = (const float*)d_in[15];
  const float* lf_bh    = (const float*)d_in[16];
  const float* lb_wi    = (const float*)d_in[17];
  const float* lb_wh    = (const float*)d_in[18];
  const float* lb_bi    = (const float*)d_in[19];
  const float* lb_bh    = (const float*)d_in[20];
  const float* ri_w     = (const float*)d_in[21];
  const float* ri_b     = (const float*)d_in[22];
  const float* adj_w    = (const float*)d_in[23];
  const float* adj_b    = (const float*)d_in[24];
  const float* inproj_w = (const float*)d_in[25];
  const float* inproj_b = (const float*)d_in[26];
  const float* outproj_w= (const float*)d_in[27];
  const float* outproj_b= (const float*)d_in[28];
  const float* lin_w    = (const float*)d_in[29];
  const float* lin_b    = (const float*)d_in[30];
  const float* f1_w     = (const float*)d_in[31];
  const float* f1_b     = (const float*)d_in[32];
  const float* f2_w     = (const float*)d_in[33];
  const float* f2_b     = (const float*)d_in[34];
  const float* g_w      = (const float*)d_in[35];
  const float* g_b      = (const float*)d_in[36];
  const int*   lengths  = (const int*)d_in[37];
  float* out = (float*)d_out;

  char* wp = (char*)d_ws;
  auto alloc = [&](size_t bytes) -> char* {
    char* p = wp;
    wp += (bytes + 255) & ~(size_t)255;
    return p;
  };

  _Float16* xs_h   = (_Float16*)alloc(BTn * 128 * sizeof(_Float16));
  float*    gi_g   = (float*)   alloc(BTn * 96  * sizeof(float));
  _Float16* li_h   = (_Float16*)alloc(BTn * 160 * sizeof(_Float16));
  _Float16* lfgi_h = (_Float16*)alloc(BTn * 256 * sizeof(_Float16));
  _Float16* lbgi_h = (_Float16*)alloc(BTn * 256 * sizeof(_Float16));
  _Float16* comb_h = (_Float16*)alloc(BTn * 128 * sizeof(_Float16));
  _Float16* xa_h   = (_Float16*)alloc(BTn * 64  * sizeof(_Float16));
  _Float16* qkv_h  = (_Float16*)alloc(BTn * 192 * sizeof(_Float16));
  _Float16* ao_h   = (_Float16*)alloc(BTn * 64  * sizeof(_Float16));
  _Float16* proj_h = (_Float16*)alloc(BTn * 64  * sizeof(_Float16));
  float*    feat   = (float*)   alloc(BTn * 64  * sizeof(float));
  float*    rnn    = (float*)   alloc(BTn * 64  * sizeof(float));
  _Float16* fcat_h = (_Float16*)alloc(BTn * 128 * sizeof(_Float16));
  _Float16* fx1_h  = (_Float16*)alloc(BTn * 64  * sizeof(_Float16));
  _Float16* fx2_h  = (_Float16*)alloc(BTn * 64  * sizeof(_Float16));
  float*    beta   = (float*)   alloc(BTn * sizeof(float));
  float*    mmn    = (float*)   alloc(Bn * 64 * 4);
  float*    rmn    = (float*)   alloc(Bn * 64 * 4);
  float*    nmeans = (float*)   alloc(Bn * 64 * 4);
  _Float16* ctx_h  = (_Float16*)alloc(Bn * 32 * 2);
  float*    h0     = (float*)   alloc(Bn * 64 * 4);
  float*    c0     = (float*)   alloc(Bn * 64 * 4);
  float*    partial= (float*)   alloc(2 * NBLK_LOSS * 4);
  float*    lfb    = (float*)   alloc(256 * 4);
  float*    lbb    = (float*)   alloc(256 * 4);
  _Float16* gru_wi_h = (_Float16*)alloc(96 * 128 * 2);
  _Float16* gru_wh_h = (_Float16*)alloc(96 * 32 * 2);
  _Float16* ih_w_h   = (_Float16*)alloc(64 * 32 * 2);
  _Float16* lf_wi_h  = (_Float16*)alloc(256 * 160 * 2);
  _Float16* lf_wh_h  = (_Float16*)alloc(256 * 64 * 2);
  _Float16* lb_wi_h  = (_Float16*)alloc(256 * 160 * 2);
  _Float16* lb_wh_h  = (_Float16*)alloc(256 * 64 * 2);
  _Float16* ri_w_h   = (_Float16*)alloc(64 * 128 * 2);
  _Float16* adj_w_h  = (_Float16*)alloc(64 * 64 * 2);
  _Float16* inpj_h   = (_Float16*)alloc(192 * 64 * 2);
  _Float16* outpj_h  = (_Float16*)alloc(64 * 64 * 2);
  _Float16* lin_w_h  = (_Float16*)alloc(64 * 64 * 2);
  _Float16* f1_w_h   = (_Float16*)alloc(64 * 128 * 2);
  _Float16* f2_w_h   = (_Float16*)alloc(64 * 64 * 2);

  auto conv = [&](const float* s, _Float16* d, int n) {
    f32_to_f16_k<<<dim3((n + 255) / 256), dim3(256), 0, stream>>>(s, d, n);
  };
  auto gemm = [&](const _Float16* A, int lda, const _Float16* W, int ldw,
                  const float* bias, float* o32, int ldo32, _Float16* o16, int ldo16,
                  long rows, int N, int K, int act) {
    if ((N & 127) == 0) {          // 256
      dim3 g((unsigned)(rows / 16), (unsigned)(N / 128));
      gemm16_k<8><<<g, dim3(32), 0, stream>>>(A, lda, W, ldw, bias, o32, ldo32, o16, ldo16, K, act);
    } else if (N % 96 == 0) {      // 96, 192
      dim3 g((unsigned)(rows / 16), (unsigned)(N / 96));
      gemm16_k<6><<<g, dim3(32), 0, stream>>>(A, lda, W, ldw, bias, o32, ldo32, o16, ldo16, K, act);
    } else {                       // 64
      dim3 g((unsigned)(rows / 16), (unsigned)(N / 64));
      gemm16_k<4><<<g, dim3(32), 0, stream>>>(A, lda, W, ldw, bias, o32, ldo32, o16, ldo16, K, act);
    }
  };

  // weight conversion + bias folding
  conv(gru_wi, gru_wi_h, 96 * 128);
  conv(gru_wh, gru_wh_h, 96 * 32);
  conv(ih_w, ih_w_h, 64 * 32);
  conv(lf_wi, lf_wi_h, 256 * 160);
  conv(lf_wh, lf_wh_h, 256 * 64);
  conv(lb_wi, lb_wi_h, 256 * 160);
  conv(lb_wh, lb_wh_h, 256 * 64);
  conv(ri_w, ri_w_h, 64 * 128);
  conv(adj_w, adj_w_h, 64 * 64);
  conv(inproj_w, inpj_h, 192 * 64);
  conv(outproj_w, outpj_h, 64 * 64);
  conv(lin_w, lin_w_h, 64 * 64);
  conv(f1_w, f1_w_h, 64 * 128);
  conv(f2_w, f2_w_h, 64 * 64);
  addvec_k<<<dim3(1), dim3(256), 0, stream>>>(lf_bi, lf_bh, lfb, 256);
  addvec_k<<<dim3(1), dim3(256), 0, stream>>>(lb_bi, lb_bh, lbb, 256);

  // stats + preprocessing
  stats_k<<<dim3(Bn), dim3(64), 0, stream>>>(values, masks, min_vals, max_vals, mmn, rmn, nmeans);
  preproc_k<<<dim3((unsigned)(Sn / 256)), dim3(256), 0, stream>>>(
      values, masks, deltas, min_vals, rmn, nmeans, decay_W, decay_b, lengths, xs_h);

  // GRU: batched input projection, then WMMA scan
  gemm(xs_h, 128, gru_wi_h, 128, gru_bi, gi_g, 96, nullptr, 0, BTn, 96, 128, 0);
  gru_scan_k<<<dim3(Bn / 16), dim3(32), 0, stream>>>(gi_g, gru_wh_h, gru_bh, lengths, ctx_h);

  // h0 / c0
  gemm(ctx_h, 32, ih_w_h, 32, ih_b, h0, 64, nullptr, 0, Bn, 64, 32, 0);
  tanh_k<<<dim3((Bn * 64) / 256), dim3(256), 0, stream>>>(h0, c0, Bn * 64);

  // LSTM inputs + scans
  build_li_k<<<dim3((unsigned)(BTn * 160 / 256)), dim3(256), 0, stream>>>(xs_h, masks, ctx_h, li_h);
  gemm(li_h, 160, lf_wi_h, 160, lfb, nullptr, 0, lfgi_h, 256, BTn, 256, 160, 0);
  gemm(li_h, 160, lb_wi_h, 160, lbb, nullptr, 0, lbgi_h, 256, BTn, 256, 160, 0);
  lstm_scan_k<<<dim3(Bn / 16), dim3(32), 0, stream>>>(lfgi_h, lf_wh_h, h0, c0, comb_h, 0);
  lstm_scan_k<<<dim3(Bn / 16), dim3(32), 0, stream>>>(lbgi_h, lb_wh_h, h0, c0, comb_h + 64, 1);

  // attention branch
  gemm(xs_h, 128, adj_w_h, 64, adj_b, nullptr, 0, xa_h, 64, BTn, 64, 64, 0);
  gemm(xa_h, 64, inpj_h, 64, inproj_b, nullptr, 0, qkv_h, 192, BTn, 192, 64, 0);
  attn_k<<<dim3(Tn / 16, NHn, Bn), dim3(32), 0, stream>>>(qkv_h, ao_h);
  gemm(ao_h, 64, outpj_h, 64, outproj_b, nullptr, 0, proj_h, 64, BTn, 64, 64, 0);
  gemm(proj_h, 64, lin_w_h, 64, lin_b, feat, 64, fcat_h + 64, 128, BTn, 64, 64, 0);

  // RNN imputation + fusion MLP
  gemm(comb_h, 128, ri_w_h, 128, ri_b, rnn, 64, fcat_h, 128, BTn, 64, 128, 0);
  gemm(fcat_h, 128, f1_w_h, 128, f1_b, nullptr, 0, fx1_h, 64, BTn, 64, 128, 1);
  gemm(fx1_h, 64, f2_w_h, 64, f2_b, nullptr, 0, fx2_h, 64, BTn, 64, 64, 1);
  beta_k<<<dim3((unsigned)(BTn / 256), 1, 1), dim3(256), 0, stream>>>(fx2_h, g_w, g_b, beta);

  // fusion + rescale + deterministic loss reduction
  final_k<<<dim3(NBLK_LOSS), dim3(256), 0, stream>>>(values, masks, min_vals, mmn, rmn, feat,
                                                     rnn, beta, lengths, out, partial);
  loss_finalize_k<<<dim3(1), dim3(256), 0, stream>>>(partial, out);
}
